// NSA_82532091560572
// MI455X (gfx1250) — matrix operand order repaired and checked by reading, not compiled, not run on hardware
//
#include <hip/hip_runtime.h>
#include <hip/hip_bf16.h>

typedef __attribute__((ext_vector_type(16))) __bf16    v16bf;
typedef __attribute__((ext_vector_type(8)))  float     v8f;
typedef __attribute__((ext_vector_type(4)))  unsigned  uint4v;

#define BX 2
#define SX 1024
#define HX 8
#define DX 128
#define NCMP 63
#define NSLC 61
#define NGRP 16
#define VTW  1312                 // 128 pad + 1024 + 160 pad
#define SCALE_CMP 11.313708499f   // reference divides by dim^-0.5
#define SCALE_INV 0.0883883476f   // dim^-0.5

#define NELEM (BX*SX*HX*DX)       // 2097152 per tensor

// ---- ws layout (BYTE offsets, all 16B aligned) ----
#define OUTCMP_B   0u
#define OUTSLC_B   (OUTCMP_B + NELEM*4u)            // + 8388608
#define PGRP_B     (OUTSLC_B + NELEM*4u)
#define PGRP_SZ    (BX*HX*NGRP*NSLC)                // 15616 floats
#define IDX_B      (PGRP_B + PGRP_SZ*4u)            // 62464 bytes of Pgrp
#define QBF_B      (IDX_B + 2048u)
#define KBF_B      (QBF_B + NELEM*2u)
#define VT_B       (KBF_B + NELEM*2u)
#define VT_SZ      (BX*HX*DX*VTW)                   // 2686976 halves
#define KCMPBF_B   (VT_B + VT_SZ*2u)
#define KCMP_SZ    (BX*HX*NCMP*DX)                  // 129024
#define VCMPT_B    (KCMPBF_B + KCMP_SZ*2u)
#define VCMPT_SZ   (BX*HX*DX*64)                    // 131072 halves

union FragAB { v16bf v; unsigned u[8]; uint4v q[2]; };

__device__ __forceinline__ unsigned short bf16bits(float x) {
  unsigned u = __float_as_uint(x);
  u += 0x7FFFu + ((u >> 16) & 1u);           // RNE
  return (unsigned short)(u >> 16);
}

// A fragment (16x32 bf16): halves 0..7 = row[kb+hi*8 + j], halves 8..15 = +16
__device__ __forceinline__ v16bf loadA(const unsigned short* __restrict__ row, int kb, int hig) {
  FragAB f;
  const unsigned short* p = row + kb + hig * 8;
  f.q[0] = *(const uint4v*)(p);
  f.q[1] = *(const uint4v*)(p + 16);
  return f.v;
}

// B fragment (32x16 bf16): halves 0..7 = row[kb+hi*8 + j], halves 8..15 = +16
__device__ __forceinline__ v16bf loadB(const unsigned short* __restrict__ row, int kb, int hig) {
  FragAB f;
  const unsigned short* p = row + kb + hig * 8;
  f.q[0] = *(const uint4v*)(p);
  f.q[1] = *(const uint4v*)(p + 16);
  return f.v;
}

#define WMMA_BF16(a, b, c) \
  __builtin_amdgcn_wmma_f32_16x16x32_bf16(false, (a), false, (b), (short)0, (c), false, false)

// ---------------- prep kernels ----------------
__global__ void k0_zero(float* __restrict__ p, int n) {
  int i = blockIdx.x * blockDim.x + threadIdx.x;
  if (i < n) p[i] = 0.f;
}

// zero the s-axis padding of vt
__global__ void kz_padvt(unsigned short* __restrict__ vt) {
  int tid = blockIdx.x * blockDim.x + threadIdx.x;
  const int total = BX * HX * DX * 288;
  if (tid >= total) return;
  int c   = tid % 288;
  int row = tid / 288;
  int col = (c < 128) ? c : (128 + SX + (c - 128));
  vt[(size_t)row * VTW + col] = 0;
}

// q,k -> bf16 (same layout)
__global__ void kc_qk(const float* __restrict__ q, const float* __restrict__ k,
                      unsigned short* __restrict__ qbf, unsigned short* __restrict__ kbf) {
  int tid = blockIdx.x * blockDim.x + threadIdx.x;
  if (tid >= NELEM) return;
  qbf[tid] = bf16bits(q[tid]);
  kbf[tid] = bf16bits(k[tid]);
}

// v -> bf16 transposed (bh, d, 128+s)
__global__ void kc_v(const float* __restrict__ v, unsigned short* __restrict__ vt) {
  int tid = blockIdx.x * blockDim.x + threadIdx.x;
  if (tid >= NELEM) return;
  int d  = tid & (DX - 1);
  int t2 = tid >> 7;
  int h  = t2 & (HX - 1);
  int t3 = t2 >> 3;
  int s  = t3 & (SX - 1);
  int b  = t3 >> 10;
  int bh = b * HX + h;
  vt[((size_t)bh * DX + d) * VTW + 128 + s] = bf16bits(v[tid]);
}

// compress K,V -> bf16 (Kcmp row-major, Vcmp transposed with padded col 63)
__global__ void k1_compress(const float* __restrict__ k, const float* __restrict__ v,
                            const float* __restrict__ wck, const float* __restrict__ bck,
                            const float* __restrict__ wcv, const float* __restrict__ bcv,
                            unsigned short* __restrict__ kcmp, unsigned short* __restrict__ vcmpt) {
  int tid = blockIdx.x * blockDim.x + threadIdx.x;
  if (tid >= KCMP_SZ) return;
  int d   = tid & (DX - 1);
  int rst = tid >> 7;
  int n   = rst % NCMP;
  int bh  = rst / NCMP;
  int h   = bh & (HX - 1);
  int b   = bh >> 3;
  size_t base = ((size_t)(b * SX + n * 16) * HX + h) * DX + d;
  float ks = 0.f, vs = 0.f;
#pragma unroll 4
  for (int l = 0; l < 32; l++) {
    ks += k[base + (size_t)l * HX * DX] * wck[l];
    vs += v[base + (size_t)l * HX * DX] * wcv[l];
  }
  kcmp[tid] = bf16bits(ks + bck[0]);
  vcmpt[((size_t)bh * DX + d) * 64 + n] = bf16bits(vs + bcv[0]);
  if (n == 0) vcmpt[((size_t)bh * DX + d) * 64 + 63] = 0;
}

// ---------------- kernel 2: compressed attention + P_slc accumulation --------
__global__ __launch_bounds__(32)
void k2_cmp(const unsigned short* __restrict__ qbf, const unsigned short* __restrict__ kcmp,
            const unsigned short* __restrict__ vcmpt, float* __restrict__ outcmp,
            float* __restrict__ Pgrp) {
  __shared__ float          Sbuf[16][64];
  __shared__ unsigned short Pbuf[16][64];
  __shared__ float          PslcS[16][NSLC];

  const int lane = threadIdx.x;
  const int m = lane & 15, hig = lane >> 4;
  const int wid = blockIdx.x;
  const int qt = wid & 63, h = (wid >> 6) & 7, b = wid >> 9;
  const int s0 = qt << 4;
  const int bh = b * HX + h;

  const unsigned short* qrow = qbf + ((size_t)(b * SX + s0 + m) * HX + h) * DX;
  const unsigned short* kcB  = kcmp + (size_t)bh * NCMP * DX;

#pragma unroll
  for (int t = 0; t < 4; t++) {
    v8f c; for (int i = 0; i < 8; i++) c[i] = 0.f;
    int n  = t * 16 + m;
    int kn = n < NCMP ? n : NCMP - 1;
    const unsigned short* krow = kcB + (size_t)kn * DX;
#pragma unroll
    for (int kb = 0; kb < DX; kb += 32)
      c = WMMA_BF16(loadA(qrow, kb, hig), loadB(krow, kb, hig), c);
#pragma unroll
    for (int r = 0; r < 8; r++) {
      int row = r + hig * 8;
      float sv = c[r] * SCALE_CMP;
      if (n >= NCMP) sv = -3.0e38f;
      Sbuf[row][t * 16 + m] = sv;
    }
  }
  __syncthreads();

  if (lane < 16) {
    const int row = lane;
    float mx = -3.0e38f;
    for (int ci = 0; ci < NCMP; ci++) mx = fmaxf(mx, Sbuf[row][ci]);
    float sum = 0.f;
    for (int ci = 0; ci < 64; ci++) {
      float e = (ci < NCMP) ? __expf(Sbuf[row][ci] - mx) : 0.f;
      Sbuf[row][ci] = e;
      sum += e;
    }
    float inv = 1.f / sum;
    for (int ci = 0; ci < 64; ci++) {
      float p = Sbuf[row][ci] * inv;
      Sbuf[row][ci] = p;
      Pbuf[row][ci] = bf16bits(p);
    }
    const float w5[5] = {1.f, 2.f, 2.f, 2.f, 1.f};
    for (int j = 0; j < NSLC; j++) {
      float a = 0.f;
      int b4 = 4 * j;
#pragma unroll
      for (int s = 0; s < 5; s++) {
        int i2 = b4 - s;
        if (i2 >= 0 && i2 < NCMP) a += w5[s] * Sbuf[row][i2];
      }
      PslcS[row][j] = a;
    }
  }
  __syncthreads();

  if (lane < 16) {
    int g = s0 >> 6;
    float* dst = Pgrp + ((size_t)bh * NGRP + g) * NSLC;
    for (int j = lane; j < NSLC; j += 16) {
      float a = 0.f;
      for (int r = 0; r < 16; r++) a += PslcS[r][j];
      atomicAdd(&dst[j], a);
    }
  }

#pragma unroll
  for (int dt = 0; dt < 8; dt++) {
    v8f o; for (int i = 0; i < 8; i++) o[i] = 0.f;
    int dcol = dt * 16 + m;
    const unsigned short* vrow = vcmpt + ((size_t)bh * DX + dcol) * 64;
#pragma unroll
    for (int kb = 0; kb < 64; kb += 32)
      o = WMMA_BF16(loadA(&Pbuf[m][0], kb, hig), loadB(vrow, kb, hig), o);
#pragma unroll
    for (int r = 0; r < 8; r++)
      outcmp[((size_t)bh * SX + s0 + r + hig * 8) * DX + dcol] = o[r];
  }
}

// ---------------- kernel 3: per-group top-2 ----------------
__global__ void k3_top2(const float* __restrict__ Pgrp, int* __restrict__ idxb) {
  int g = blockIdx.x * blockDim.x + threadIdx.x;
  if (g >= BX * HX * NGRP) return;
  const float* row = Pgrp + (size_t)g * NSLC;
  int i0 = 0; float v0 = row[0];
  for (int j = 1; j < NSLC; j++) { float x = row[j]; if (x > v0) { v0 = x; i0 = j; } }
  int i1 = (i0 == 0) ? 1 : 0; float v1 = row[i1];
  for (int j = 0; j < NSLC; j++) {
    if (j == i0) continue;
    float x = row[j];
    if (x > v1) { v1 = x; i1 = j; }
  }
  idxb[2 * g] = i0; idxb[2 * g + 1] = i1;
}

// ---------------- kernel 4: selected-block attention ----------------
__global__ __launch_bounds__(32)
void k4_slc(const unsigned short* __restrict__ qbf, const unsigned short* __restrict__ kbf,
            const unsigned short* __restrict__ vt, const int* __restrict__ idxb,
            float* __restrict__ outslc) {
  __shared__ float          Sbuf[16][128];
  __shared__ unsigned short Pbuf[16][128];

  const int lane = threadIdx.x;
  const int m = lane & 15, hig = lane >> 4;
  const int wid = blockIdx.x;
  const int qt = wid & 63, h = (wid >> 6) & 7, b = wid >> 9;
  const int s0 = qt << 4;
  const int bh = b * HX + h;
  const int g = s0 >> 6;

  const int blk0 = idxb[((size_t)bh * NGRP + g) * 2 + 0];
  const int blk1 = idxb[((size_t)bh * NGRP + g) * 2 + 1];

  const unsigned short* qrow = qbf + ((size_t)(b * SX + s0 + m) * HX + h) * DX;

#pragma unroll
  for (int t = 0; t < 8; t++) {
    v8f c; for (int i = 0; i < 8; i++) c[i] = 0.f;
    int kk  = t * 16 + m;
    int blk = (kk >> 6) ? blk1 : blk0;
    int srw = blk * 16 + (kk & 63);
    const unsigned short* krow = kbf + ((size_t)(b * SX + srw) * HX + h) * DX;
#pragma unroll
    for (int kb = 0; kb < DX; kb += 32)
      c = WMMA_BF16(loadA(qrow, kb, hig), loadB(krow, kb, hig), c);
#pragma unroll
    for (int r = 0; r < 8; r++)
      Sbuf[r + hig * 8][t * 16 + m] = c[r] * SCALE_INV;
  }
  __syncthreads();

  if (lane < 16) {
    const int row = lane;
    float mx = -3.0e38f;
    for (int ci = 0; ci < 128; ci++) mx = fmaxf(mx, Sbuf[row][ci]);
    float sum = 0.f;
    for (int ci = 0; ci < 128; ci++) { float e = __expf(Sbuf[row][ci] - mx); Sbuf[row][ci] = e; sum += e; }
    float inv = 1.f / sum;
    for (int ci = 0; ci < 128; ci++) Pbuf[row][ci] = bf16bits(Sbuf[row][ci] * inv);
  }
  __syncthreads();

#pragma unroll
  for (int dt = 0; dt < 8; dt++) {
    v8f o; for (int i = 0; i < 8; i++) o[i] = 0.f;
    int dcol = dt * 16 + m;
    const unsigned short* vrow = vt + ((size_t)bh * DX + dcol) * VTW + 128;
#pragma unroll
    for (int kb = 0; kb < 128; kb += 32) {
      int blk = (kb >> 6) ? blk1 : blk0;
      o = WMMA_BF16(loadA(&Pbuf[m][0], kb, hig),
                    loadB(vrow + blk * 16 + (kb & 63), 0, hig), o);
    }
#pragma unroll
    for (int r = 0; r < 8; r++)
      outslc[((size_t)bh * SX + s0 + r + hig * 8) * DX + dcol] = o[r];
  }
}

// ---------------- kernel 5: window attention + gate + combine ----------------
__global__ __launch_bounds__(32)
void k5_win(const float* __restrict__ q, const unsigned short* __restrict__ qbf,
            const unsigned short* __restrict__ kbf, const unsigned short* __restrict__ vt,
            const float* __restrict__ wg, const float* __restrict__ bg,
            const float* __restrict__ outcmp, const float* __restrict__ outslc,
            float* __restrict__ out) {
  __shared__ float          Sbuf[16][288];
  __shared__ unsigned short Pbuf[16][288];
  __shared__ float          gbuf[16][3];

  const int lane = threadIdx.x;
  const int m = lane & 15, hig = lane >> 4;
  const int wid = blockIdx.x;
  const int qt = wid & 63, h = (wid >> 6) & 7, b = wid >> 9;
  const int s0 = qt << 4;
  const int bh = b * HX + h;
  const int kstart = s0 - 128;

  const unsigned short* qrow = qbf + ((size_t)(b * SX + s0 + m) * HX + h) * DX;

  if (lane < 16) {
    const float* qr = q + ((size_t)(b * SX + s0 + lane) * HX + h) * DX;
#pragma unroll
    for (int c = 0; c < 3; c++) {
      float acc = 0.f;
      for (int d = 0; d < DX; d++) acc += qr[d] * wg[c * DX + d];
      gbuf[lane][c] = 1.f / (1.f + __expf(-(acc + bg[c])));
    }
  }

#pragma unroll 2
  for (int t = 0; t < 18; t++) {
    v8f c; for (int i = 0; i < 8; i++) c[i] = 0.f;
    int j  = kstart + t * 16 + m;
    int jc = j < 0 ? 0 : (j > SX - 1 ? SX - 1 : j);
    const unsigned short* krow = kbf + ((size_t)(b * SX + jc) * HX + h) * DX;
#pragma unroll
    for (int kb = 0; kb < DX; kb += 32)
      c = WMMA_BF16(loadA(qrow, kb, hig), loadB(krow, kb, hig), c);
#pragma unroll
    for (int r = 0; r < 8; r++) {
      int row = r + hig * 8;
      int i   = s0 + row;
      bool ok = (j >= 0) && (j < SX) && (j >= i - 128) && (j <= i + 128);
      Sbuf[row][t * 16 + m] = ok ? c[r] * SCALE_INV : -3.0e38f;
    }
  }
  __syncthreads();

  if (lane < 16) {
    const int row = lane;
    float mx = -3.0e38f;
    for (int ci = 0; ci < 288; ci++) mx = fmaxf(mx, Sbuf[row][ci]);
    float sum = 0.f;
    for (int ci = 0; ci < 288; ci++) { float e = __expf(Sbuf[row][ci] - mx); Sbuf[row][ci] = e; sum += e; }
    float inv = 1.f / sum;
    for (int ci = 0; ci < 288; ci++) Pbuf[row][ci] = bf16bits(Sbuf[row][ci] * inv);
  }
  __syncthreads();

#pragma unroll
  for (int dt = 0; dt < 8; dt++) {
    v8f o; for (int i = 0; i < 8; i++) o[i] = 0.f;
    int dcol = dt * 16 + m;
    const unsigned short* vrow = vt + ((size_t)bh * DX + dcol) * VTW + 128 + kstart;
#pragma unroll
    for (int kb = 0; kb < 288; kb += 32)
      o = WMMA_BF16(loadA(&Pbuf[m][0], kb, hig), loadB(vrow, kb, hig), o);
#pragma unroll
    for (int r = 0; r < 8; r++) {
      int row  = r + hig * 8;
      int si   = s0 + row;
      size_t wsidx = ((size_t)bh * SX + si) * DX + dcol;
      float res = gbuf[row][0] * outcmp[wsidx] + gbuf[row][1] * outslc[wsidx] + gbuf[row][2] * o[r];
      out[((size_t)(b * SX + si) * HX + h) * DX + dcol] = res;
    }
  }
}

extern "C" void kernel_launch(void* const* d_in, const int* in_sizes, int n_in,
                              void* d_out, int out_size, void* d_ws, size_t ws_size,
                              hipStream_t stream) {
  (void)in_sizes; (void)n_in; (void)out_size; (void)ws_size;
  const float* q   = (const float*)d_in[0];
  const float* k   = (const float*)d_in[1];
  const float* v   = (const float*)d_in[2];
  const float* wck = (const float*)d_in[3];
  const float* bck = (const float*)d_in[4];
  const float* wcv = (const float*)d_in[5];
  const float* bcv = (const float*)d_in[6];
  const float* wg  = (const float*)d_in[7];
  const float* bg  = (const float*)d_in[8];
  float* out = (float*)d_out;
  char*  ws  = (char*)d_ws;

  float* outcmp = (float*)(ws + OUTCMP_B);
  float* outslc = (float*)(ws + OUTSLC_B);
  float* Pgrp   = (float*)(ws + PGRP_B);
  int*   idxb   = (int*)(ws + IDX_B);
  unsigned short* qbf   = (unsigned short*)(ws + QBF_B);
  unsigned short* kbf   = (unsigned short*)(ws + KBF_B);
  unsigned short* vt    = (unsigned short*)(ws + VT_B);
  unsigned short* kcmpb = (unsigned short*)(ws + KCMPBF_B);
  unsigned short* vcmpt = (unsigned short*)(ws + VCMPT_B);

  const int nwaves = BX * HX * (SX / 16);   // 1024 16-query tiles
  const int padN   = BX * HX * DX * 288;

  k0_zero<<<(PGRP_SZ + 255) / 256, 256, 0, stream>>>(Pgrp, PGRP_SZ);
  kz_padvt<<<(padN + 255) / 256, 256, 0, stream>>>(vt);
  kc_qk<<<(NELEM + 255) / 256, 256, 0, stream>>>(q, k, qbf, kbf);
  kc_v<<<(NELEM + 255) / 256, 256, 0, stream>>>(v, vt);
  k1_compress<<<(KCMP_SZ + 255) / 256, 256, 0, stream>>>(k, v, wck, bck, wcv, bcv, kcmpb, vcmpt);
  k2_cmp<<<nwaves, 32, 0, stream>>>(qbf, kcmpb, vcmpt, outcmp, Pgrp);
  k3_top2<<<1, 256, 0, stream>>>(Pgrp, idxb);
  k4_slc<<<nwaves, 32, 0, stream>>>(qbf, kbf, vt, idxb, outslc);
  k5_win<<<nwaves, 32, 0, stream>>>(q, qbf, kbf, vt, wg, bg, outcmp, outslc, out);
}
